// DIS_84877143704250
// MI455X (gfx1250) — matrix operand order repaired and checked by reading, not compiled, not run on hardware
//
#include <hip/hip_runtime.h>
#include <hip/hip_bf16.h>
#include <math.h>

// Problem constants
#define BB   16384
#define ZZ   128
#define HH   512
#define HIDD 512
#define TT   32
#define HL2PI 0.91893853320467274f   // 0.5*log(2*pi)

typedef __bf16        v16bf __attribute__((ext_vector_type(16)));
typedef float         v8f   __attribute__((ext_vector_type(8)));
typedef unsigned int  v4u   __attribute__((ext_vector_type(4)));

union Frag { v4u q[2]; v16bf v; };

__device__ __forceinline__ unsigned short f2bf(float f) {
  unsigned int u = __float_as_uint(f);
  u += 0x7FFFu + ((u >> 16) & 1u);          // round-to-nearest-even
  return (unsigned short)(u >> 16);
}

// A-fragment (16x32, bf16) from row-major [rows][ldk] buffer.
// p points at element (row_for_lane, kbase) where kbase = 32*kc + (lane>>4)*8.
// Holds K = kbase..kbase+7 and kbase+16..kbase+23 per ISA layout.
__device__ __forceinline__ v16bf afrag(const unsigned short* p) {
  Frag f;
  f.q[0] = *(const v4u*)(p);
  f.q[1] = *(const v4u*)(p + 16);
  return f.v;
}
// B-fragment (32x16, bf16) from N-major (transposed) weight [N][ldk].
// p points at element (n_for_lane, kbase2) where kbase2 = 32*kc + (lane>>4)*8;
// Holds K = kbase2..kbase2+7 and kbase2+16..kbase2+23 (same as A).
__device__ __forceinline__ v16bf bfrag(const unsigned short* p) {
  Frag f;
  f.q[0] = *(const v4u*)(p);
  f.q[1] = *(const v4u*)(p + 16);
  return f.v;
}

// ---------------------------------------------------------------------------
// Kernel 1: transpose + convert weights to bf16 (N-major for B-operands)
// ---------------------------------------------------------------------------
__global__ void prep_weights(const float* __restrict__ W1, const float* __restrict__ W2,
                             unsigned short* __restrict__ w1zt,   // [512][128]
                             unsigned short* __restrict__ w1ct,   // [512][512]
                             unsigned short* __restrict__ w2t) {  // [128][512]
  int i = blockIdx.x * blockDim.x + threadIdx.x;
  if (i < 512 * 128) { int n = i >> 7, k = i & 127; w1zt[i] = f2bf(W1[k * 512 + n]); }
  if (i < 512 * 512) { int n = i >> 9, k = i & 511; w1ct[i] = f2bf(W1[(128 + k) * 512 + n]); }
  if (i < 128 * 512) { int n = i >> 9, k = i & 511; w2t[i]  = f2bf(W2[k * 128 + n]); }
}

// ---------------------------------------------------------------------------
// Kernel 2: context -> bf16
// ---------------------------------------------------------------------------
__global__ void conv_ctx(const float* __restrict__ ctx, unsigned short* __restrict__ ctxbf) {
  int i = blockIdx.x * blockDim.x + threadIdx.x;
  if (i < BB * HH) ctxbf[i] = f2bf(ctx[i]);
}

// ---------------------------------------------------------------------------
// Kernel 3: a[b][n] = ctx[b]@W1c[:,n] + b1[n]   (WMMA, one 16x16 tile / wave)
// ---------------------------------------------------------------------------
__global__ void precompute_a(const unsigned short* __restrict__ ctxbf,  // [B][512]
                             const unsigned short* __restrict__ w1ct,   // [512][512] N-major
                             const float* __restrict__ b1,
                             float* __restrict__ a_out) {               // [B][512]
  int lane = threadIdx.x & 31;
  int w    = threadIdx.x >> 5;            // 0..3
  int rt   = blockIdx.x;                  // 0..1023 row tile
  int nt   = blockIdx.y * 4 + w;          // 0..31 col tile
  int llo = lane & 15, lhi = lane >> 4;

  int rowA = rt * 16 + llo;
  int kA   = lhi * 8;
  int nB   = nt * 16 + llo;
  int kB   = lhi * 8;

  v8f acc = {};
#pragma unroll
  for (int kc = 0; kc < 16; ++kc) {
    v16bf af = afrag(ctxbf + (size_t)rowA * 512 + kc * 32 + kA);
    v16bf bf = bfrag(w1ct  + (size_t)nB   * 512 + kc * 32 + kB);
    acc = __builtin_amdgcn_wmma_f32_16x16x32_bf16(false, af, false, bf,
                                                  (short)0, acc, false, false);
  }
  int col = nt * 16 + llo;
  float bias = b1[col];
#pragma unroll
  for (int r = 0; r < 8; ++r) {
    int row = rt * 16 + r + 8 * lhi;
    a_out[(size_t)row * 512 + col] = acc[r] + bias;
  }
}

// ---------------------------------------------------------------------------
// Kernel 4: main T-step loop. One workgroup (16 waves) owns 32 batch rows,
// weights resident in LDS for the whole trajectory. Weights are staged with
// the CDNA5 direct global->LDS async path (ASYNCcnt).
// ---------------------------------------------------------------------------
__global__ __launch_bounds__(512)
void dis_main(const unsigned short* __restrict__ w1zt,  // [512][128] bf16 N-major
              const unsigned short* __restrict__ w2t,   // [128][512] bf16 N-major
              const float* __restrict__ a_glob,         // [B][512]
              const float* __restrict__ eps0,           // [B][Z]
              const float* __restrict__ eps,            // [T][B][Z]
              const float* __restrict__ beta,           // [T]
              const float* __restrict__ sigma0,
              const float* __restrict__ b2,             // [Z]
              const float* __restrict__ t_emb,          // [T][512]
              const float* __restrict__ target_mu,      // [B][Z]
              float* __restrict__ out) {
  extern __shared__ char smem[];
  // LDS layout (byte offsets within this workgroup's allocation, base = 0):
  unsigned short* s_w1  = (unsigned short*)(smem);           // [  0..131071] 512*128 bf16
  unsigned short* s_w2  = (unsigned short*)(smem + 131072);  // 128*512 bf16
  unsigned short* s_zb  = (unsigned short*)(smem + 262144);  // 32*128 bf16
  float*          s_zf  = (float*)        (smem + 270336);   // 32*128 f32
  unsigned short* s_hid = (unsigned short*)(smem + 286720);  // 32*512 bf16

  int tid  = threadIdx.x;
  int lane = tid & 31;
  int w    = tid >> 5;                 // wave 0..15
  int row0 = blockIdx.x * 32;          // global batch-row base
  float sg0 = sigma0[0];
  const float dt = 1.0f / (float)TT;

  // ---- async DMA weights -> LDS (global_load_async_to_lds_b128, ASYNCcnt) ----
  // 2 x 8192 16-byte chunks; per-lane LDS byte offset rides in a VGPR.
  for (int i = tid; i < 8192; i += 512) {
    unsigned           l1 = (unsigned)(i * 16);                 // s_w1 region base 0
    unsigned long long g1 = (unsigned long long)w1zt + (unsigned long long)i * 16u;
    asm volatile("global_load_async_to_lds_b128 %0, %1, off"
                 :: "v"(l1), "v"(g1) : "memory");
    unsigned           l2 = (unsigned)(131072 + i * 16);        // s_w2 region
    unsigned long long g2 = (unsigned long long)w2t + (unsigned long long)i * 16u;
    asm volatile("global_load_async_to_lds_b128 %0, %1, off"
                 :: "v"(l2), "v"(g2) : "memory");
  }

  // overlap: init z0 = sigma0 * eps0 while the DMA runs
  for (int i = tid; i < 32 * 128; i += 512) {
    int rl = i >> 7, c = i & 127;
    float z0 = sg0 * eps0[(size_t)(row0 + rl) * 128 + c];
    s_zf[i] = z0;
    s_zb[i] = f2bf(z0);
  }

  // wave tiling: GEMM1 -> (mi, nt = ntb+8j, j=0..3); GEMM2 -> (mi, ntb)
  int mi  = w & 1;
  int ntb = w >> 1;
  int llo = lane & 15, lhi = lane >> 4;
  int rowA = mi * 16 + llo;     // A-frag row (local, both GEMMs)
  int kA   = lhi * 8;
  int kB   = lhi * 8;

  // persistent per-tile "a" (ctx@W1c + b1) registers
  float areg[4][8];
#pragma unroll
  for (int j = 0; j < 4; ++j) {
    int col = (ntb + 8 * j) * 16 + llo;
#pragma unroll
    for (int r = 0; r < 8; ++r) {
      int row = row0 + mi * 16 + r + 8 * lhi;
      areg[j][r] = a_glob[(size_t)row * 512 + col];
    }
  }
  int   col2 = ntb * 16 + llo;          // this wave's Z column (GEMM2 output)
  float b2r  = b2[col2];

  // log_w init: -log N(z0; 0, sigma0) = 0.5*eps0^2 + log(sigma0) + 0.5log2pi
  float lw[8];
  float lgs0 = __logf(sg0);
#pragma unroll
  for (int r = 0; r < 8; ++r) {
    int rowl = mi * 16 + r + 8 * lhi;
    float e0 = eps0[(size_t)(row0 + rowl) * 128 + col2];
    lw[r] = 0.5f * e0 * e0 + lgs0 + HL2PI;
  }

  // drain this wave's async DMAs; barrier makes all waves' weight chunks visible
  asm volatile("s_wait_asynccnt 0x0" ::: "memory");

  for (int t = 0; t < TT; ++t) {
    float bf_ = beta[t];
    float bb_ = beta[(t + TT - 1) & (TT - 1)];
    float sigf = sqrtf(2.0f * bf_ * dt) * sg0;
    float sigb = sqrtf(2.0f * bb_ * dt) * sg0;
    float isb  = 1.0f / sigb;
    float lgr  = 0.5f * (__logf(bf_) - __logf(bb_));   // log(sig_f) - log(sig_b)

    __syncthreads();   // z tile (and, on t==0, weights) ready

    // ---- GEMM1: hid_pre[32,512] = z[32,128] @ W1z ----
    v8f acc1[4] = {};
#pragma unroll
    for (int kc = 0; kc < 4; ++kc) {
      v16bf af = afrag(s_zb + rowA * 128 + kc * 32 + kA);
#pragma unroll
      for (int j = 0; j < 4; ++j) {
        int nt = ntb + 8 * j;
        v16bf bg = bfrag(s_w1 + (nt * 16 + llo) * 128 + kc * 32 + kB);
        acc1[j] = __builtin_amdgcn_wmma_f32_16x16x32_bf16(false, af, false, bg,
                                                          (short)0, acc1[j], false, false);
      }
    }
    // epilogue: + a + t_emb, ReLU, -> LDS bf16
#pragma unroll
    for (int j = 0; j < 4; ++j) {
      int col = (ntb + 8 * j) * 16 + llo;
      float te = t_emb[t * 512 + col];
#pragma unroll
      for (int r = 0; r < 8; ++r) {
        int rowl = mi * 16 + r + 8 * lhi;
        float h = acc1[j][r] + areg[j][r] + te;
        h = fmaxf(h, 0.0f);
        s_hid[rowl * 512 + col] = f2bf(h);
      }
    }
    __syncthreads();   // hid tile ready

    // ---- GEMM2: u[32,128] = hid[32,512] @ W2 ----
    v8f acc2 = {};
#pragma unroll
    for (int kc = 0; kc < 16; ++kc) {
      v16bf af = afrag(s_hid + rowA * 512 + kc * 32 + kA);
      v16bf bg = bfrag(s_w2 + col2 * 512 + kc * 32 + kB);
      acc2 = __builtin_amdgcn_wmma_f32_16x16x32_bf16(false, af, false, bg,
                                                     (short)0, acc2, false, false);
    }

    // ---- elementwise SDE step + importance weight ----
#pragma unroll
    for (int r = 0; r < 8; ++r) {
      int rowl = mi * 16 + r + 8 * lhi;
      float u   = acc2[r] + b2r;
      float zp  = s_zf[rowl * 128 + col2];
      float muf = zp + (bf_ * zp + u) * dt;
      float e   = eps[(size_t)t * (BB * ZZ) + (size_t)(row0 + rowl) * 128 + col2];
      float zn  = muf + sigf * e;                  // (zn - muf)/sigf == e exactly
      float mub = zn - bb_ * zn * dt;
      float d   = (zp - mub) * isb;
      lw[r] += 0.5f * e * e + lgr - 0.5f * d * d;
      s_zf[rowl * 128 + col2] = zn;
      s_zb[rowl * 128 + col2] = f2bf(zn);
    }
    if (t < TT - 1)
      __builtin_prefetch(eps + (size_t)(t + 1) * (BB * ZZ) + (size_t)(row0 + mi * 16) * 128 + col2, 0, 1);
  }

  // terminal: + log N(zT; target_mu, 1)
#pragma unroll
  for (int r = 0; r < 8; ++r) {
    int rowl = mi * 16 + r + 8 * lhi;
    float zT = s_zf[rowl * 128 + col2];
    float dd = zT - target_mu[(size_t)(row0 + rowl) * 128 + col2];
    lw[r] += -0.5f * dd * dd - HL2PI;
  }
  float s = 0.0f;
#pragma unroll
  for (int r = 0; r < 8; ++r) s += lw[r];
  for (int off = 16; off > 0; off >>= 1) s += __shfl_down(s, off, 32);
  if (lane == 0) atomicAdd(out, s * (1.0f / (float)BB));
}

// ---------------------------------------------------------------------------
extern "C" void kernel_launch(void* const* d_in, const int* in_sizes, int n_in,
                              void* d_out, int out_size, void* d_ws, size_t ws_size,
                              hipStream_t stream) {
  const float* ctx  = (const float*)d_in[0];
  const float* eps0 = (const float*)d_in[1];
  const float* eps  = (const float*)d_in[2];
  const float* beta = (const float*)d_in[3];
  const float* sg0  = (const float*)d_in[4];
  const float* W1   = (const float*)d_in[5];
  const float* b1   = (const float*)d_in[6];
  const float* W2   = (const float*)d_in[7];
  const float* b2   = (const float*)d_in[8];
  const float* temb = (const float*)d_in[9];
  const float* tmu  = (const float*)d_in[10];
  float* out = (float*)d_out;

  char* ws = (char*)d_ws;
  float*          a_g   = (float*)ws;                          // 33,554,432 B
  unsigned short* ctxbf = (unsigned short*)(ws + 33554432);    // 16,777,216 B
  unsigned short* w1zt  = (unsigned short*)(ws + 50331648);    //    131,072 B
  unsigned short* w1ct  = (unsigned short*)(ws + 50462720);    //    524,288 B
  unsigned short* w2t   = (unsigned short*)(ws + 50987008);    //    131,072 B

  hipMemsetAsync(d_out, 0, sizeof(float) * out_size, stream);
  prep_weights<<<1024, 256, 0, stream>>>(W1, W2, w1zt, w1ct, w2t);
  conv_ctx<<<(BB * HH) / 256, 256, 0, stream>>>(ctx, ctxbf);
  precompute_a<<<dim3(BB / 16, 8), 128, 0, stream>>>(ctxbf, w1ct, b1, a_g);
  dis_main<<<BB / 32, 512, 319488, stream>>>(w1zt, w2t, a_g, eps0, eps, beta,
                                             sg0, b2, temb, tmu, out);
}
